// DynamicSpectralPolarityAttention_18365280157942
// MI455X (gfx1250) — matrix operand order repaired and checked by reading, not compiled, not run on hardware
//
#include <hip/hip_runtime.h>
#include <hip/hip_bf16.h>
#include <math.h>

typedef __attribute__((ext_vector_type(16))) _Float16 v16h;
typedef __attribute__((ext_vector_type(8)))  _Float16 v8h;
typedef __attribute__((ext_vector_type(8)))  float    v8f;

#define PIF 3.14159265358979f

// ---------------- workspace layout (bytes, 256-aligned) ----------------
constexpr size_t AL(size_t x){ return (x + 255) & ~(size_t)255; }
constexpr size_t OFF_XH    = 0;                                   // x f16: 8*1024*512
constexpr size_t OFF_WQKV  = OFF_XH    + AL(4194304ull*2);        // qkv_w f16: 3*1536*512
constexpr size_t OFF_WMI   = OFF_WQKV  + AL(2359296ull*2);        // mha_in_w f16
constexpr size_t OFF_WMO   = OFF_WMI   + AL(786432ull*2);         // mha_out_w f16
constexpr size_t OFF_WPROJ = OFF_WMO   + AL(262144ull*2);         // proj_w f16
constexpr size_t OFF_WIHF  = OFF_WPROJ + AL(262144ull*2);         // lstm wih_f f16
constexpr size_t OFF_WIHB  = OFF_WIHF  + AL(262144ull*2);         // lstm wih_b f16
constexpr size_t OFF_GXF   = OFF_WIHB  + AL(262144ull*2);         // gates fwd f32 (8192,512)
constexpr size_t OFF_GXB   = OFF_GXF   + AL(4194304ull*4);
constexpr size_t OFF_LCTX  = OFF_GXB   + AL(4194304ull*4);        // (8,256)
constexpr size_t OFF_QKVC  = OFF_LCTX  + AL(2048ull*4);           // mha qkv f32 (8192,1536)
constexpr size_t OFF_QKVCH = OFF_QKVC  + AL(12582912ull*4);       // f16 copy
constexpr size_t OFF_SCORES= OFF_QKVCH + AL(12582912ull*2);       // (32,1024,1024) f32
constexpr size_t OFF_PROBS = OFF_SCORES+ AL(33554432ull*4);       // f16 probs
constexpr size_t OFF_ATTNO = OFF_PROBS + AL(33554432ull*2);       // (8192,512) f32
constexpr size_t OFF_ATTNOH= OFF_ATTNO + AL(4194304ull*4);
constexpr size_t OFF_ACOUT = OFF_ATTNOH+ AL(4194304ull*2);        // (8192,512) f32
constexpr size_t OFF_XMEAN = OFF_ACOUT + AL(4194304ull*4);        // (8,512)
constexpr size_t OFF_ACMEAN= OFF_XMEAN + AL(4096ull*4);
constexpr size_t OFF_PARAMS= OFF_ACMEAN+ AL(4096ull*4);           // coeffs/gmix/smix/mw
constexpr size_t OFF_QKVS  = OFF_PARAMS+ AL(1024ull*4);           // per-scale qkv (8192,1536)
constexpr size_t OFF_U     = OFF_QKVS  + AL(12582912ull*4);       // (128,16,1024) cplx
constexpr size_t OFF_TW    = OFF_U     + AL(4194304ull*4);        // (128,16,16) cplx
constexpr size_t OFF_QSPEC = OFF_TW    + AL(65536ull*4);          // (64,1024,64) f32
constexpr size_t OFF_KSPEC = OFF_QSPEC + AL(4194304ull*4);
constexpr size_t OFF_YS    = OFF_KSPEC + AL(4194304ull*4);        // (8,1024,512) per-scale
constexpr size_t OFF_YACC  = OFF_YS    + AL(4194304ull*4);        // accumulated over scales
constexpr size_t OFF_YH    = OFF_YACC  + AL(4194304ull*4);        // f16 copy of yacc
constexpr size_t OFF_VT    = OFF_YH    + AL(4194304ull*2);        // v^T f16 (32,128,1024)

// ---------------- f32 -> f16 conversion ----------------
__global__ void k_cvt_h(const float* __restrict__ src, _Float16* __restrict__ dst, long n){
  long i = (long)blockIdx.x*256 + threadIdx.x;
  long stride = (long)gridDim.x*256;
  for (long j = i; j < n; j += stride) dst[j] = (_Float16)src[j];
}

// ---------------- transpose V heads for attn@V: vT[(b*4+h2),d,m] ------------
__global__ void k_transpose_v(const _Float16* __restrict__ qkvch, _Float16* __restrict__ vT){
  int gid = blockIdx.x*256 + threadIdx.x;       // 32*128*1024 = 4,194,304
  if (gid >= 32*128*1024) return;
  int m = gid & 1023;
  int d = (gid >> 10) & 127;
  int inst = gid >> 17;                         // b*4 + h2
  int b = inst >> 2, h2 = inst & 3;
  vT[gid] = qkvch[((size_t)(b*1024 + m)*1536) + 1024 + h2*128 + d];
}

// ---------------- WMMA fragment loaders (wave32 ISA layouts) ----------------
// A 16x32 f16: lane<16 -> row=lane, K {0..7, 16..23}; lane>=16 -> row=lane-16,
// K {8..15, 24..31}. Elements of v16h are K-ascending within each half.
__device__ __forceinline__ v16h load_afrag(const _Float16* __restrict__ arow, int off){
  v16h af;
  const v8h a0 = *(const v8h*)(arow + off);
  const v8h a1 = *(const v8h*)(arow + off + 16);
  #pragma unroll
  for (int j = 0; j < 8; ++j){ af[j] = a0[j]; af[j+8] = a1[j]; }
  return af;
}
// B 32x16 f16 (from W stored (N,K) row-major): lane<16 -> col=lane, K {0..7, 16..23};
// lane>=16 -> col=lane-16, K {8..15, 24..31}.
__device__ __forceinline__ v16h load_bfrag(const _Float16* __restrict__ bp){
  v16h bf;
  const v8h b0 = *(const v8h*)bp;
  const v8h b1 = *(const v8h*)(bp + 16);
  #pragma unroll
  for (int j = 0; j < 8; ++j){ bf[j] = b0[j]; bf[j+8] = b1[j]; }
  return bf;
}

// ---------------- WMMA GEMM: C = alpha*(A @ B^T) + bias + bias2 -------------
// A: (M,K) f16 row-major lda.  B: (N,K) f16 row-major ldb.
// Block = 8 waves as 4x2, each wave owns a 32x32 C tile -> block tile 128x64.
// K must be a multiple of 64. Batched via blockIdx.z -> (z/zdiv, z%zdiv).
__global__ __launch_bounds__(256)
void k_gemm(const _Float16* __restrict__ A, const _Float16* __restrict__ B,
            float* __restrict__ C, const float* __restrict__ bias,
            const float* __restrict__ bias2,
            int M, int N, int K, int lda, int ldb, int ldc,
            long sA0, long sA1, long sB0, long sB1, long sC0, long sC1,
            int zdiv, float alpha)
{
  int z = blockIdx.z, z0 = z / zdiv, z1 = z - z0 * zdiv;
  A += (size_t)z0*sA0 + (size_t)z1*sA1;
  B += (size_t)z0*sB0 + (size_t)z1*sB1;
  C += (size_t)z0*sC0 + (size_t)z1*sC1;
  int w = threadIdx.x >> 5, lane = threadIdx.x & 31;
  int l = lane & 15, hi = lane >> 4;
  int mbase = blockIdx.y*128 + (w & 3)*32;
  int nbase = blockIdx.x*64  + (w >> 2)*32;
  if (mbase >= M || nbase >= N) return;
  v8f acc00 = {}, acc01 = {}, acc10 = {}, acc11 = {};
  const int kbA = hi*8, kbB = hi*8;
  const _Float16* arow0 = A + (size_t)(mbase + l)*lda;
  const _Float16* arow1 = A + (size_t)(mbase + 16 + l)*lda;
  const _Float16* brow0 = B + (size_t)(nbase + l)*ldb;
  const _Float16* brow1 = B + (size_t)(nbase + 16 + l)*ldb;
  for (int kk = 0; kk < K; kk += 64) {
    __builtin_prefetch(arow0 + kk + 256, 0, 1);   // global_prefetch on A stream
    #pragma unroll
    for (int u2 = 0; u2 < 2; ++u2) {
      const int ko = kk + u2*32;
      v16h a0 = load_afrag(arow0, ko + kbA);
      v16h a1 = load_afrag(arow1, ko + kbA);
      v16h b0 = load_bfrag(brow0 + ko + kbB);
      v16h b1 = load_bfrag(brow1 + ko + kbB);
      acc00 = __builtin_amdgcn_wmma_f32_16x16x32_f16(false, a0, false, b0, (short)0, acc00, false, false);
      acc01 = __builtin_amdgcn_wmma_f32_16x16x32_f16(false, a0, false, b1, (short)0, acc01, false, false);
      acc10 = __builtin_amdgcn_wmma_f32_16x16x32_f16(false, a1, false, b0, (short)0, acc10, false, false);
      acc11 = __builtin_amdgcn_wmma_f32_16x16x32_f16(false, a1, false, b1, (short)0, acc11, false, false);
    }
  }
  // C tile store: lane covers col l (+0/+16), rows hi*8 + v (+0/+16)
  int ncol0 = nbase + l, ncol1 = nbase + 16 + l;
  float badd0 = (bias ? bias[ncol0] : 0.f) + (bias2 ? bias2[ncol0] : 0.f);
  float badd1 = (bias ? bias[ncol1] : 0.f) + (bias2 ? bias2[ncol1] : 0.f);
  int mrow0 = mbase + hi*8, mrow1 = mbase + 16 + hi*8;
  #pragma unroll
  for (int v = 0; v < 8; ++v){
    C[(size_t)(mrow0 + v)*ldc + ncol0] = acc00[v]*alpha + badd0;
    C[(size_t)(mrow0 + v)*ldc + ncol1] = acc01[v]*alpha + badd1;
    C[(size_t)(mrow1 + v)*ldc + ncol0] = acc10[v]*alpha + badd0;
    C[(size_t)(mrow1 + v)*ldc + ncol1] = acc11[v]*alpha + badd1;
  }
}

// ---------------- mean over N of (B,N,C) ----------------
__global__ void k_mean_rows(const float* __restrict__ in, float* __restrict__ out,
                            int Bn, int Nn, int Cc){
  int idx = blockIdx.x*256 + threadIdx.x;
  if (idx >= Bn*Cc) return;
  int b = idx / Cc, c = idx - b*Cc;
  const float* p = in + (size_t)b*Nn*Cc + c;
  float s = 0.f;
  for (int n = 0; n < Nn; ++n) s += p[(size_t)n*Cc];
  out[idx] = s / (float)Nn;
}

// ---------------- row softmax (f32 in, f16 out) ----------------
__global__ __launch_bounds__(256)
void k_softmax_row(const float* __restrict__ in, _Float16* __restrict__ out, int L){
  __shared__ float red[256];
  const float* row = in + (size_t)blockIdx.x*L;
  _Float16* orow = out + (size_t)blockIdx.x*L;
  int t = threadIdx.x;
  float m = -3.4e38f;
  for (int i = t; i < L; i += 256) m = fmaxf(m, row[i]);
  red[t] = m; __syncthreads();
  for (int s = 128; s > 0; s >>= 1){ if (t < s) red[t] = fmaxf(red[t], red[t+s]); __syncthreads(); }
  m = red[0]; __syncthreads();
  float sum = 0.f;
  for (int i = t; i < L; i += 256) sum += __expf(row[i]-m);
  red[t] = sum; __syncthreads();
  for (int s = 128; s > 0; s >>= 1){ if (t < s) red[t] += red[t+s]; __syncthreads(); }
  float inv = 1.f/red[0];
  for (int i = t; i < L; i += 256) orow[i] = (_Float16)(__expf(row[i]-m)*inv);
}

// ---------------- sequential LSTM scan (one block per direction) ------------
__global__ __launch_bounds__(256)
void k_lstm(const float* __restrict__ gx, const float* __restrict__ whh,
            float* __restrict__ lctx, int dir){
  __shared__ float h[8][128], c[8][128], hsum[8][128], gsh[8][512];
  int t = threadIdx.x;
  for (int i = t; i < 1024; i += 256){ int b = i>>7, m = i&127; h[b][m]=0.f; c[b][m]=0.f; hsum[b][m]=0.f; }
  __syncthreads();
  for (int step = 0; step < 1024; ++step){
    int n = dir ? (1023 - step) : step;
    for (int q = 0; q < 16; ++q){
      int o = t*16 + q;          // o = b*512 + j
      int b = o >> 9, j = o & 511;
      float acc = gx[((size_t)b*1024 + n)*512 + j];
      const float* wr = whh + (size_t)j*128;
      #pragma unroll 4
      for (int m = 0; m < 128; ++m) acc += h[b][m]*wr[m];
      gsh[b][j] = acc;
    }
    __syncthreads();
    for (int q = 0; q < 4; ++q){
      int o = t*4 + q; int b = o>>7, m = o&127;
      float gi = gsh[b][m], gf = gsh[b][128+m], gg = gsh[b][256+m], go = gsh[b][384+m];
      float ci = (1.f/(1.f+__expf(-gf)))*c[b][m] + (1.f/(1.f+__expf(-gi)))*tanhf(gg);
      float hn = (1.f/(1.f+__expf(-go)))*tanhf(ci);
      c[b][m] = ci; h[b][m] = hn; hsum[b][m] += hn;
    }
    __syncthreads();
  }
  for (int i = t; i < 1024; i += 256){ int b = i>>7, m = i&127;
    lctx[b*256 + dir*128 + m] = hsum[b][m]*(1.f/1024.f); }
}

// ---------------- context head: coeffs, gmix, smix, mode weights ------------
// params (floats): [0..255] coeffs(b*32+h*4+j), [256..271] gmix(h*2+j),
//                  [272..279] smix(h), [280..282] mw
__global__ __launch_bounds__(256)
void k_head(const float* __restrict__ xmean, const float* __restrict__ acmean,
            const float* __restrict__ lctx,
            const float* __restrict__ ca_w1, const float* __restrict__ ca_b1,
            const float* __restrict__ ca_w2, const float* __restrict__ ca_b2,
            const float* __restrict__ ge_b1, const float* __restrict__ ge_w2,
            const float* __restrict__ ge_b2,
            const float* __restrict__ spec_mix, const float* __restrict__ mode_w,
            float* __restrict__ params){
  __shared__ float comb[8][512];
  __shared__ float t1s[8][256];
  __shared__ float eg[16];
  int t = threadIdx.x;
  for (int i = t; i < 4096; i += 256){ int b = i>>9, cc = i&511;
    float lc = (cc < 256) ? lctx[b*256 + cc] : 0.f;
    comb[b][cc] = xmean[b*512+cc] + acmean[b*512+cc] + lc; }
  __syncthreads();
  for (int o = t; o < 2048; o += 256){ int b = o>>8, i = o&255;
    float s = ca_b1[i]; const float* wr = ca_w1 + (size_t)i*512;
    for (int c2 = 0; c2 < 512; ++c2) s += comb[b][c2]*wr[c2];
    t1s[b][i] = fmaxf(s, 0.f); }
  __syncthreads();
  if (t < 256){ int b = t>>5, hj = t&31;
    float s = ca_b2[hj]; const float* wr = ca_w2 + (size_t)hj*256;
    for (int i = 0; i < 256; ++i) s += t1s[b][i]*wr[i];
    params[t] = 1.f/(1.f+__expf(-s)); }
  if (t < 16){ float s = ge_b2[t]; const float* wr = ge_w2 + (size_t)t*256;
    for (int i = 0; i < 256; ++i) s += tanhf(ge_b1[i])*wr[i];
    eg[t] = s; }
  __syncthreads();
  if (t < 8){ float a = eg[t*2], b2 = eg[t*2+1]; float m = fmaxf(a, b2);
    float ea = __expf(a-m), ebv = __expf(b2-m), inv = 1.f/(ea+ebv);
    params[256+t*2] = ea*inv; params[256+t*2+1] = ebv*inv; }
  if (t < 8){ float m = -3.0e38f;
    for (int k = 0; k < 16; ++k) m = fmaxf(m, spec_mix[t*16+k]);
    float se = 0.f; for (int k = 0; k < 16; ++k) se += __expf(spec_mix[t*16+k]-m);
    float mean = 0.f; for (int k = 0; k < 16; ++k) mean += __expf(spec_mix[t*16+k]-m)/se;
    params[272+t] = mean/16.f; }
  if (t == 0){ float m = fmaxf(mode_w[0], fmaxf(mode_w[1], mode_w[2]));
    float e0 = __expf(mode_w[0]-m), e1 = __expf(mode_w[1]-m), e2 = __expf(mode_w[2]-m);
    float inv = 1.f/(e0+e1+e2);
    params[280] = e0*inv; params[281] = e1*inv; params[282] = e2*inv; }
}

// ---------------- spectral stage 1: DFT over hd (64 -> 16 freqs) ------------
__global__ __launch_bounds__(256)
void k_spec_fwd1(const float* __restrict__ qkvs, float* __restrict__ u){
  __shared__ float tabc[64], tabs[64];
  int t = threadIdx.x;
  if (t < 64){ float a = -2.f*PIF*(float)t/64.f; tabc[t] = __cosf(a); tabs[t] = __sinf(a); }
  __syncthreads();
  int inst = blockIdx.x;
  int bh = inst >> 1, which = inst & 1;
  int b = bh >> 3, h = bh & 7;
  int n = blockIdx.y*16 + (t >> 4);
  int k2 = t & 15;
  const float* trow = qkvs + ((size_t)(b*1024 + n)*1536) + which*512 + h*64;
  float sr = 0.f, si = 0.f;
  for (int d = 0; d < 64; ++d){
    int m = (k2*d) & 63;
    float xv = trow[d];
    sr += xv*tabc[m]; si += xv*tabs[m];
  }
  size_t o = (((size_t)inst*16 + k2)*1024 + n)*2;
  u[o] = sr; u[o+1] = si;
}

// ---------------- spectral stage 2: DFT over N + complex weight -------------
__global__ __launch_bounds__(256)
void k_spec_fwd2(const float* __restrict__ u, const float* __restrict__ pos_sw,
                 const float* __restrict__ neg_sw, const float* __restrict__ params,
                 float* __restrict__ tw){
  __shared__ float tc[1024], ts[1024];
  int t = threadIdx.x;
  for (int i = t; i < 1024; i += 256){ float a = -2.f*PIF*(float)i/1024.f;
    tc[i] = __cosf(a); ts[i] = __sinf(a); }
  __syncthreads();
  int inst = blockIdx.x;
  int h = (inst >> 1) & 7;
  int k1 = t >> 4, k2 = t & 15;
  const float* up = u + ((size_t)inst*16 + k2)*1024*2;
  float sr = 0.f, si = 0.f;
  for (int n = 0; n < 1024; ++n){
    int m = (k1*n) & 1023;
    float ur = up[2*n], ui = up[2*n+1];
    float cv = tc[m], sv = ts[m];
    sr += ur*cv - ui*sv;
    si += ur*sv + ui*cv;
  }
  float smix = params[272+h];
  size_t wi = (((size_t)h*16 + k1)*64 + k2)*2;
  float wr  = pos_sw[wi]   + smix*neg_sw[wi];
  float wim = pos_sw[wi+1] + smix*neg_sw[wi+1];
  size_t o = ((size_t)inst*256 + t)*2;
  tw[o]   = sr*wr - si*wim;
  tw[o+1] = sr*wim + si*wr;
}

// ---------------- spectral stage 3: inverse (16x16 -> 1024x64 real) ---------
__global__ __launch_bounds__(256)
void k_spec_inv(const float* __restrict__ tw, float* __restrict__ qspec,
                float* __restrict__ kspec){
  __shared__ float twr[16][16], twi[16][16], dc[16][64], dsn[16][64];
  int t = threadIdx.x;
  int inst = blockIdx.x; int bh = inst >> 1; int which = inst & 1;
  twr[t>>4][t&15] = tw[((size_t)inst*256 + t)*2];
  twi[t>>4][t&15] = tw[((size_t)inst*256 + t)*2 + 1];
  for (int i = t; i < 1024; i += 256){ int k2 = i>>6, d = i&63;
    float a = 2.f*PIF*(float)(k2*d)/64.f; dc[k2][d] = __cosf(a); dsn[k2][d] = __sinf(a); }
  __syncthreads();
  int n = blockIdx.y*256 + t;
  float vr[16], vi[16];
  #pragma unroll
  for (int k2 = 0; k2 < 16; ++k2){ vr[k2] = 0.f; vi[k2] = 0.f; }
  for (int k1 = 0; k1 < 16; ++k1){
    float a = 2.f*PIF*(float)((k1*n) & 1023)/1024.f;
    float cv = __cosf(a), sv = __sinf(a);
    #pragma unroll
    for (int k2 = 0; k2 < 16; ++k2){
      vr[k2] += twr[k1][k2]*cv - twi[k1][k2]*sv;
      vi[k2] += twr[k1][k2]*sv + twi[k1][k2]*cv;
    }
  }
  float* outp = (which ? kspec : qspec) + ((size_t)bh*1024 + n)*64;
  const float scale = 1.f/(1024.f*64.f);
  for (int d = 0; d < 64; ++d){
    float s = vr[0];
    #pragma unroll
    for (int k2 = 1; k2 < 16; ++k2) s += 2.f*(vr[k2]*dc[k2][d] - vi[k2]*dsn[k2][d]);
    outp[d] = s*scale;
  }
}

// ---------------- polarity linear attention (one block per (b,h)) -----------
__global__ __launch_bounds__(256)
void k_polarity(const float* __restrict__ qspec, const float* __restrict__ kspec,
                const float* __restrict__ qkvs, const float* __restrict__ spw,
                const float* __restrict__ params, float* __restrict__ ys, int s){
  __shared__ float kp[64][64], kn[64][64], vv[64][64];
  __shared__ float pv[64], ksp[64], ksn[64];
  __shared__ float qp_s[4][64], qn_s[4][64];
  int t = threadIdx.x;
  int bh = blockIdx.x; int b = bh >> 3, h = bh & 7;
  if (t < 64) pv[t] = 1.f + fabsf(spw[((size_t)s*8 + h)*64 + t]);
  __syncthreads();
  int d = t >> 2, eg = (t & 3)*16;
  float accp[16], accn[16];
  #pragma unroll
  for (int j = 0; j < 16; ++j){ accp[j] = 0.f; accn[j] = 0.f; }
  float kspa = 0.f, ksna = 0.f;
  for (int ch = 0; ch < 16; ++ch){
    for (int q = 0; q < 16; ++q){
      int f = t*16 + q; int r = f >> 6, ci = f & 63;
      float kvv = kspec[(((size_t)bh*1024) + ch*64 + r)*64 + ci];
      float p = pv[ci];
      kp[r][ci] = (kvv > 0.f) ? __powf(kvv, p) : 0.f;
      kn[r][ci] = (kvv < 0.f) ? __powf(-kvv, p) : 0.f;
      vv[r][ci] = qkvs[((size_t)(b*1024 + ch*64 + r)*1536) + 1024 + h*64 + ci];
    }
    __syncthreads();
    for (int r = 0; r < 64; ++r){
      float kpv = kp[r][d], knv = kn[r][d];
      if (eg == 0){ kspa += kpv; ksna += knv; }
      float* vvr = &vv[r][eg];
      #pragma unroll
      for (int j = 0; j < 16; ++j){ float vj = vvr[j]; accp[j] += kpv*vj; accn[j] += knv*vj; }
    }
    __syncthreads();
  }
  if (eg == 0){ ksp[d] = kspa; ksn[d] = ksna; }
  __syncthreads();
  #pragma unroll
  for (int j = 0; j < 16; ++j){ kp[d][eg+j] = accp[j]; kn[d][eg+j] = accn[j]; }
  __syncthreads();
  float c0 = params[(b*8+h)*4+0], c1 = params[(b*8+h)*4+1];
  float c2 = params[(b*8+h)*4+2], c3 = params[(b*8+h)*4+3];
  float g0 = params[256 + h*2], g1 = params[256 + h*2 + 1];
  for (int nb = 0; nb < 1024; nb += 4){
    { int r = t >> 6, dd = t & 63;
      float qv = qspec[(((size_t)bh*1024) + nb + r)*64 + dd];
      float p = pv[dd];
      qp_s[r][dd] = (qv > 0.f) ? __powf(qv, p) : 0.f;
      qn_s[r][dd] = (qv < 0.f) ? __powf(-qv, p) : 0.f; }
    __syncthreads();
    int r = t >> 6, e = t & 63;
    float app = 0.f, ann = 0.f, apn = 0.f, anp = 0.f, z = 1e-6f;
    for (int dd = 0; dd < 64; ++dd){
      float qp1 = qp_s[r][dd], qn1 = qn_s[r][dd];
      app += qp1*kp[dd][e]; ann += qn1*kn[dd][e];
      apn += qp1*kn[dd][e]; anp += qn1*kp[dd][e];
      z += qp1*ksp[dd] + qn1*ksn[dd];
    }
    ys[((size_t)(b*1024 + nb + r)*512) + h*64 + e] =
        (g0*(c0*app + c1*ann) + g1*(c2*apn + c3*anp))/z;
    __syncthreads();
  }
}

// ---------------- depthwise 5/7/9 conv mix, += into ys ----------------------
__global__ void k_dwc(const float* __restrict__ qkvs,
                      const float* __restrict__ w5, const float* __restrict__ b5,
                      const float* __restrict__ w7, const float* __restrict__ b7,
                      const float* __restrict__ w9, const float* __restrict__ b9,
                      const float* __restrict__ params, float* __restrict__ ys){
  int gid = blockIdx.x*256 + threadIdx.x;
  if (gid >= 8*8*1024*64) return;
  int e = gid & 63;
  int n = (gid >> 6) & 1023;
  int h = (gid >> 16) & 7;
  int b = gid >> 19;
  int y = n >> 5, x = n & 31;
  float mw0 = params[280], mw1 = params[281], mw2 = params[282];
  const size_t vbase = ((size_t)b*1024)*1536 + 1024 + h*64 + e;
  float acc = mw0*b5[e] + mw1*b7[e] + mw2*b9[e];
  for (int ky = 0; ky < 5; ++ky){ int yy = y + ky - 2; if (yy < 0 || yy > 31) continue;
    for (int kx = 0; kx < 5; ++kx){ int xx = x + kx - 2; if (xx < 0 || xx > 31) continue;
      acc += mw0*w5[e*25 + ky*5 + kx]*qkvs[vbase + (size_t)(yy*32 + xx)*1536]; }}
  for (int ky = 0; ky < 7; ++ky){ int yy = y + ky - 3; if (yy < 0 || yy > 31) continue;
    for (int kx = 0; kx < 7; ++kx){ int xx = x + kx - 3; if (xx < 0 || xx > 31) continue;
      acc += mw1*w7[e*49 + ky*7 + kx]*qkvs[vbase + (size_t)(yy*32 + xx)*1536]; }}
  for (int ky = 0; ky < 9; ++ky){ int yy = y + ky - 4; if (yy < 0 || yy > 31) continue;
    for (int kx = 0; kx < 9; ++kx){ int xx = x + kx - 4; if (xx < 0 || xx > 31) continue;
      acc += mw2*w9[e*81 + ky*9 + kx]*qkvs[vbase + (size_t)(yy*32 + xx)*1536]; }}
  ys[((size_t)(b*1024 + n)*512) + h*64 + e] += acc;
}

// ---------------- accumulate scale contribution -----------------------------
__global__ void k_accum(const float* __restrict__ ys, float* __restrict__ yacc,
                        int n, int first){
  int i = blockIdx.x*256 + threadIdx.x;
  if (i >= n) return;
  float v = ys[i]*(1.f/3.f);
  yacc[i] = first ? v : (yacc[i] + v);
}

// ---------------- host side -------------------------------------------------
static void cvt(hipStream_t st, const float* s, _Float16* d, long n){
  long blk = (n + 255)/256; if (blk > 16384) blk = 16384; if (blk < 1) blk = 1;
  k_cvt_h<<<dim3((unsigned)blk), dim3(256), 0, st>>>(s, d, n);
}

static void gemm(hipStream_t st, const _Float16* A, const _Float16* B, float* C,
                 const float* bias, const float* bias2,
                 int M, int N, int K, int lda, int ldb, int ldc,
                 long sA0, long sA1, long sB0, long sB1, long sC0, long sC1,
                 int nz, int zdiv, float alpha){
  dim3 g(N/64, M/128, nz);
  k_gemm<<<g, dim3(256), 0, st>>>(A, B, C, bias, bias2, M, N, K, lda, ldb, ldc,
                                  sA0, sA1, sB0, sB1, sC0, sC1, zdiv, alpha);
}

extern "C" void kernel_launch(void* const* d_in, const int* in_sizes, int n_in,
                              void* d_out, int out_size, void* d_ws, size_t ws_size,
                              hipStream_t stream){
  (void)in_sizes; (void)n_in; (void)out_size; (void)ws_size;
  const float* x        = (const float*)d_in[0];
  const float* qkv_w    = (const float*)d_in[3];
  const float* proj_w   = (const float*)d_in[4];
  const float* proj_b   = (const float*)d_in[5];
  const float* pos_sw   = (const float*)d_in[6];
  const float* neg_sw   = (const float*)d_in[7];
  const float* spec_mix = (const float*)d_in[8];
  const float* ca_w1    = (const float*)d_in[9];
  const float* ca_b1    = (const float*)d_in[10];
  const float* ca_w2    = (const float*)d_in[11];
  const float* ca_b2    = (const float*)d_in[12];
  const float* wih_f    = (const float*)d_in[13];
  const float* whh_f    = (const float*)d_in[14];
  const float* bih_f    = (const float*)d_in[15];
  const float* bhh_f    = (const float*)d_in[16];
  const float* wih_b    = (const float*)d_in[17];
  const float* whh_b    = (const float*)d_in[18];
  const float* bih_b    = (const float*)d_in[19];
  const float* bhh_b    = (const float*)d_in[20];
  const float* mha_in_w = (const float*)d_in[21];
  const float* mha_in_b = (const float*)d_in[22];
  const float* mha_out_w= (const float*)d_in[23];
  const float* mha_out_b= (const float*)d_in[24];
  const float* ge_b1    = (const float*)d_in[26];
  const float* ge_w2    = (const float*)d_in[27];
  const float* ge_b2    = (const float*)d_in[28];
  const float* spw      = (const float*)d_in[29];
  const float* w5 = (const float*)d_in[30]; const float* b5 = (const float*)d_in[31];
  const float* w7 = (const float*)d_in[32]; const float* b7 = (const float*)d_in[33];
  const float* w9 = (const float*)d_in[34]; const float* b9 = (const float*)d_in[35];
  const float* mode_w   = (const float*)d_in[36];

  char* ws = (char*)d_ws;
  _Float16* xh     = (_Float16*)(ws + OFF_XH);
  _Float16* wqkvh  = (_Float16*)(ws + OFF_WQKV);
  _Float16* wmih   = (_Float16*)(ws + OFF_WMI);
  _Float16* wmoh   = (_Float16*)(ws + OFF_WMO);
  _Float16* wprojh = (_Float16*)(ws + OFF_WPROJ);
  _Float16* wihfh  = (_Float16*)(ws + OFF_WIHF);
  _Float16* wihbh  = (_Float16*)(ws + OFF_WIHB);
  float*    gxf    = (float*)(ws + OFF_GXF);
  float*    gxb    = (float*)(ws + OFF_GXB);
  float*    lctx   = (float*)(ws + OFF_LCTX);
  float*    qkvc   = (float*)(ws + OFF_QKVC);
  _Float16* qkvch  = (_Float16*)(ws + OFF_QKVCH);
  float*    scores = (float*)(ws + OFF_SCORES);
  _Float16* probs  = (_Float16*)(ws + OFF_PROBS);
  float*    attno  = (float*)(ws + OFF_ATTNO);
  _Float16* attnoh = (_Float16*)(ws + OFF_ATTNOH);
  float*    acout  = (float*)(ws + OFF_ACOUT);
  float*    xmean  = (float*)(ws + OFF_XMEAN);
  float*    acmean = (float*)(ws + OFF_ACMEAN);
  float*    params = (float*)(ws + OFF_PARAMS);
  float*    qkvs   = (float*)(ws + OFF_QKVS);
  float*    u      = (float*)(ws + OFF_U);
  float*    tw     = (float*)(ws + OFF_TW);
  float*    qspec  = (float*)(ws + OFF_QSPEC);
  float*    kspec  = (float*)(ws + OFF_KSPEC);
  float*    ysb    = (float*)(ws + OFF_YS);
  float*    yacc   = (float*)(ws + OFF_YACC);
  _Float16* yh     = (_Float16*)(ws + OFF_YH);
  _Float16* vT     = (_Float16*)(ws + OFF_VT);

  // f16 conversions
  cvt(stream, x, xh, 8L*1024*512);
  cvt(stream, qkv_w, wqkvh, 3L*1536*512);
  cvt(stream, mha_in_w, wmih, 1536L*512);
  cvt(stream, mha_out_w, wmoh, 512L*512);
  cvt(stream, proj_w, wprojh, 512L*512);
  cvt(stream, wih_f, wihfh, 512L*512);
  cvt(stream, wih_b, wihbh, 512L*512);

  // LSTM input gates (WMMA) then sequential scan
  gemm(stream, xh, wihfh, gxf, bih_f, bhh_f, 8192, 512, 512, 512, 512, 512,
       0,0,0,0,0,0, 1, 1, 1.f);
  gemm(stream, xh, wihbh, gxb, bih_b, bhh_b, 8192, 512, 512, 512, 512, 512,
       0,0,0,0,0,0, 1, 1, 1.f);
  k_lstm<<<dim3(1), dim3(256), 0, stream>>>(gxf, whh_f, lctx, 0);
  k_lstm<<<dim3(1), dim3(256), 0, stream>>>(gxb, whh_b, lctx, 1);

  // MHA branch
  gemm(stream, xh, wmih, qkvc, mha_in_b, nullptr, 8192, 1536, 512, 512, 512, 1536,
       0,0,0,0,0,0, 1, 1, 1.f);
  cvt(stream, qkvc, qkvch, 8192L*1536);
  k_transpose_v<<<dim3(16384), dim3(256), 0, stream>>>(qkvch, vT);
  // scores: z = b*4 + h2
  gemm(stream, qkvch, qkvch + 512, scores, nullptr, nullptr,
       1024, 1024, 128, 1536, 1536, 1024,
       1024L*1536, 128, 1024L*1536, 128, 4L*1024*1024, 1024L*1024,
       32, 4, 1.f/sqrtf(128.f));
  k_softmax_row<<<dim3(32*1024), dim3(256), 0, stream>>>(scores, probs, 1024);
  // attn @ v via vT (B rows are v columns, contiguous in K)
  gemm(stream, probs, vT, attno, nullptr, nullptr,
       1024, 128, 1024, 1024, 1024, 512,
       4L*1024*1024, 1024L*1024, 4L*128*1024, 128L*1024, 1024L*512, 128,
       32, 4, 1.f);
  cvt(stream, attno, attnoh, 8192L*512);
  gemm(stream, attnoh, wmoh, acout, mha_out_b, nullptr, 8192, 512, 512, 512, 512, 512,
       0,0,0,0,0,0, 1, 1, 1.f);

  // means + head
  k_mean_rows<<<dim3(16), dim3(256), 0, stream>>>(x, xmean, 8, 1024, 512);
  k_mean_rows<<<dim3(16), dim3(256), 0, stream>>>(acout, acmean, 8, 1024, 512);
  k_head<<<dim3(1), dim3(256), 0, stream>>>(xmean, acmean, lctx, ca_w1, ca_b1,
      ca_w2, ca_b2, ge_b1, ge_w2, ge_b2, spec_mix, mode_w, params);

  // three scales
  for (int s = 0; s < 3; ++s){
    gemm(stream, xh, wqkvh + (size_t)s*1536*512, qkvs, nullptr, nullptr,
         8192, 1536, 512, 512, 512, 1536, 0,0,0,0,0,0, 1, 1, 1.f);
    k_spec_fwd1<<<dim3(128, 64), dim3(256), 0, stream>>>(qkvs, u);
    k_spec_fwd2<<<dim3(128), dim3(256), 0, stream>>>(u, pos_sw, neg_sw, params, tw);
    k_spec_inv<<<dim3(128, 4), dim3(256), 0, stream>>>(tw, qspec, kspec);
    k_polarity<<<dim3(64), dim3(256), 0, stream>>>(qspec, kspec, qkvs, spw, params, ysb, s);
    k_dwc<<<dim3(16384), dim3(256), 0, stream>>>(qkvs, w5, b5, w7, b7, w9, b9, params, ysb);
    k_accum<<<dim3(16384), dim3(256), 0, stream>>>(ysb, yacc, 8*1024*512, s == 0 ? 1 : 0);
  }

  // final projection into d_out
  cvt(stream, yacc, yh, 8192L*512);
  gemm(stream, yh, wprojh, (float*)d_out, proj_b, nullptr, 8192, 512, 512, 512, 512, 512,
       0,0,0,0,0,0, 1, 1, 1.f);
}
